// SimilarityBasedImportance_7610682048676
// MI455X (gfx1250) — compile-verified, matrix operand order checked by reading, not run on hardware
//
#include <hip/hip_runtime.h>

typedef __attribute__((ext_vector_type(16))) __bf16 v16bf;
typedef __attribute__((ext_vector_type(8)))  __bf16 v8bf;
typedef __attribute__((ext_vector_type(8)))  float  v8f;

#define D_IN  256
#define D_ENC 128

// ---------------------------------------------------------------------------
// CDNA5 async global->LDS copy (16B per lane, tracked on ASYNCcnt).
// vdst = LDS byte address (low 32 bits of the generic pointer ARE the LDS
// offset per ISA 10.2 aperture mapping), vaddr = 64-bit global address.
// ---------------------------------------------------------------------------
__device__ inline void async_cp16(void* lds, const void* g)
{
    unsigned lo = (unsigned)(uintptr_t)lds;
    asm volatile("global_load_async_to_lds_b128 %0, %1, off"
                 :: "v"(lo), "v"(g) : "memory");
}
__device__ inline void wait_async0()
{
    asm volatile("s_wait_asynccnt 0" ::: "memory");
}

// ---------------------------------------------------------------------------
// Phase 1: m = X @ W + b (f32 compute), store bf16 + per-row squared norms of
// the bf16-rounded values (keeps d2 self-consistent with the bf16 WMMA).
// ---------------------------------------------------------------------------
__global__ __launch_bounds__(256)
void encode_kernel(const float* __restrict__ X, const float* __restrict__ W,
                   const float* __restrict__ bias, __bf16* __restrict__ out,
                   float* __restrict__ norms)
{
    __shared__ float Wlds[128 * D_ENC];   // 64 KB

    const int tid  = threadIdx.x;
    const int lane = tid & 31;
    const int wave = tid >> 5;
    const int rowBase = blockIdx.x * 32;

    float acc[4][4];
    #pragma unroll
    for (int r = 0; r < 4; ++r)
        #pragma unroll
        for (int j = 0; j < 4; ++j) acc[r][j] = 0.0f;

    for (int pass = 0; pass < 2; ++pass) {
        for (int i = tid; i < 128 * D_ENC; i += 256)
            Wlds[i] = W[(size_t)(pass * 128) * D_ENC + i];
        __syncthreads();

        #pragma unroll
        for (int r = 0; r < 4; ++r) {
            const int row = rowBase + wave * 4 + r;
            const float4 xc = *(const float4*)(X + (size_t)row * D_IN + pass * 128 + lane * 4);
            float x4[4] = {xc.x, xc.y, xc.z, xc.w};
            for (int s = 0; s < 32; ++s) {
                #pragma unroll
                for (int e = 0; e < 4; ++e) {
                    const float xk = __shfl(x4[e], s, 32);
                    const int   k  = s * 4 + e;
                    #pragma unroll
                    for (int j = 0; j < 4; ++j)
                        acc[r][j] = fmaf(xk, Wlds[k * D_ENC + lane + 32 * j], acc[r][j]);
                }
            }
        }
        __syncthreads();
    }

    #pragma unroll
    for (int r = 0; r < 4; ++r) {
        const int row = rowBase + wave * 4 + r;
        float ss = 0.0f;
        #pragma unroll
        for (int j = 0; j < 4; ++j) {
            const int c = lane + 32 * j;
            float v = acc[r][j] + bias[c];
            __bf16 bv = (__bf16)v;
            out[(size_t)row * D_ENC + c] = bv;
            float fv = (float)bv;
            ss = fmaf(fv, fv, ss);
        }
        #pragma unroll
        for (int m = 16; m >= 1; m >>= 1)
            ss += __shfl_xor(ss, m, 32);
        if (lane == 0) norms[row] = ss;
    }
}

// ---------------------------------------------------------------------------
// Fragment loader (LDS): CDNA5 16-bit A/B operand layout. Lane L holds row
// (M or N) = L&15; 16B chunk at K = k0+hi*8, second chunk at K = k0+16+hi*8.
// ---------------------------------------------------------------------------
__device__ inline v16bf frag_lds(const __bf16* p)
{
    v8bf lo = *(const v8bf*)(p);
    v8bf hv = *(const v8bf*)(p + 16);
    v16bf f;
    #pragma unroll
    for (int i = 0; i < 8; ++i) { f[i] = lo[i]; f[i + 8] = hv[i]; }
    return f;
}

#define LDS_STRIDE 136   // 128 + 8 halves pad (272B = 17*16B): kills bank-0 alias

// ---------------------------------------------------------------------------
// Phase 2: fused S = m @ r^T -> d2 = ||m||^2+||r||^2-2S -> row/col MIN of d2.
// sim = 1/(1+sqrt(eps+d2)) is monotone decreasing in d2, so max(sim) is
// computed from min(d2) by a single final transform (saves 268M sqrt+rcp).
// Block tile 256x128 staged in LDS via async-to-LDS; 8 waves as 4(M) x 2(N);
// wave tile 64x64 = 4x4 WMMA accumulators, K=128 in 4 steps.
// ---------------------------------------------------------------------------
__global__ __launch_bounds__(256)
void pairwise_kernel(const __bf16* __restrict__ mbf, const __bf16* __restrict__ rbf,
                     const float* __restrict__ normM, const float* __restrict__ normR,
                     float* __restrict__ outM, float* __restrict__ outR)
{
    __shared__ __bf16 mT[256][LDS_STRIDE];   // 68 KB
    __shared__ __bf16 rT[128][LDS_STRIDE];   // 34 KB
    __shared__ float  sNormM[256];
    __shared__ float  sNormR[128];
    __shared__ int    sRowMin[256];
    __shared__ int    sColMin[128];

    const int tid  = threadIdx.x;
    const int lane = tid & 31;
    const int wave = tid >> 5;
    const int wm   = wave >> 1;        // 0..3  (M direction, 64 rows each)
    const int wn   = wave & 1;         // 0..1  (N direction, 64 cols each)
    const int hi   = lane >> 4;        // 0/1
    const int ln   = lane & 15;

    const int rowBlock = blockIdx.y * 256;
    const int colBlock = blockIdx.x * 128;
    const int INFB = 0x7f800000;

    sNormM[tid]  = normM[rowBlock + tid];
    sRowMin[tid] = INFB;
    if (tid < 128) {
        sNormR[tid]  = normR[colBlock + tid];
        sColMin[tid] = INFB;
    }

    // ---- async stage of m-tile (256x128) and r-tile (128x128) into LDS ----
    {
        const char* gm = (const char*)(mbf + (size_t)rowBlock * D_ENC);
        for (int c = tid; c < 4096; c += 256) {          // 4096 x 16B = 64KB
            const int row = c >> 4, wi = c & 15;
            async_cp16(&mT[row][wi * 8], gm + row * 256 + wi * 16);
        }
        const char* gr = (const char*)(rbf + (size_t)colBlock * D_ENC);
        for (int c = tid; c < 2048; c += 256) {          // 2048 x 16B = 32KB
            const int row = c >> 4, wi = c & 15;
            async_cp16(&rT[row][wi * 8], gr + row * 256 + wi * 16);
        }
    }
    wait_async0();
    __syncthreads();

    // ---- WMMA main loop (fragments from LDS) ----
    v8f zero = {};
    v8f acc[4][4];
    #pragma unroll
    for (int mt = 0; mt < 4; ++mt)
        #pragma unroll
        for (int nt = 0; nt < 4; ++nt) acc[mt][nt] = zero;

    #pragma unroll
    for (int k0 = 0; k0 < D_ENC; k0 += 32) {
        v16bf b[4];
        #pragma unroll
        for (int nt = 0; nt < 4; ++nt)
            b[nt] = frag_lds(&rT[wn * 64 + nt * 16 + ln][k0 + hi * 8]);
        #pragma unroll
        for (int mt = 0; mt < 4; ++mt) {
            v16bf a = frag_lds(&mT[wm * 64 + mt * 16 + ln][k0 + hi * 8]);
            #pragma unroll
            for (int nt = 0; nt < 4; ++nt)
                acc[mt][nt] = __builtin_amdgcn_wmma_f32_16x16x32_bf16(
                    false, a, false, b[nt], (short)0, acc[mt][nt], false, false);
        }
    }

    // ---- fused epilogue: d2 + row/col min (pure VALU, no transcendentals) --
    const float FINF = __int_as_float(INFB);
    float colPart[4] = {FINF, FINF, FINF, FINF};

    #pragma unroll
    for (int mt = 0; mt < 4; ++mt) {
        #pragma unroll
        for (int i = 0; i < 8; ++i) {
            const int   mloc = wm * 64 + mt * 16 + hi * 8 + i;   // C layout: M = i + hi*8
            const float nm   = sNormM[mloc];
            float rowPart = FINF;
            #pragma unroll
            for (int nt = 0; nt < 4; ++nt) {
                const int   nloc = wn * 64 + nt * 16 + ln;        // C layout: N = lane&15
                const float d2   = fmaxf(fmaf(-2.0f, acc[mt][nt][i], nm + sNormR[nloc]), 0.0f);
                rowPart     = fminf(rowPart, d2);
                colPart[nt] = fminf(colPart[nt], d2);
            }
            #pragma unroll
            for (int m = 8; m >= 1; m >>= 1)
                rowPart = fminf(rowPart, __shfl_xor(rowPart, m, 16));
            if (ln == 0)
                atomicMin(&sRowMin[mloc], __float_as_int(rowPart));
        }
    }
    #pragma unroll
    for (int nt = 0; nt < 4; ++nt) {
        float c = fminf(colPart[nt], __shfl_xor(colPart[nt], 16, 32));
        if (hi == 0)
            atomicMin(&sColMin[wn * 64 + nt * 16 + ln], __float_as_int(c));
    }
    __syncthreads();

    // nonneg floats: int-bit compare == float compare -> exact, deterministic
    for (int t = tid; t < 384; t += 256) {
        if (t < 256) atomicMin((int*)&outM[rowBlock + t], sRowMin[t]);
        else         atomicMin((int*)&outR[colBlock + t - 256], sColMin[t - 256]);
    }
}

__global__ void init_out_kernel(float* __restrict__ out, int n)
{
    int i = blockIdx.x * blockDim.x + threadIdx.x;
    if (i < n) out[i] = __int_as_float(0x7f800000);   // +inf (min identity)
}

__global__ void finalize_kernel(float* __restrict__ out, int n)
{
    int i = blockIdx.x * blockDim.x + threadIdx.x;
    if (i < n) {
        float d2 = out[i];
        out[i] = 1.0f / (1.0f + sqrtf(1e-5f + d2));
    }
}

// ---------------------------------------------------------------------------
extern "C" void kernel_launch(void* const* d_in, const int* in_sizes, int n_in,
                              void* d_out, int out_size, void* d_ws, size_t ws_size,
                              hipStream_t stream)
{
    const float* match = (const float*)d_in[0];
    const float* refer = (const float*)d_in[1];
    const float* Wm    = (const float*)d_in[2];
    const float* bm    = (const float*)d_in[3];
    const float* Wr    = (const float*)d_in[4];
    const float* br    = (const float*)d_in[5];

    const int n1 = in_sizes[0] / D_IN;   // 16384
    const int n2 = in_sizes[1] / D_IN;   // 16384

    char* ws = (char*)d_ws;
    __bf16* mbf   = (__bf16*)ws;                                    // n1*128*2 B
    __bf16* rbf   = (__bf16*)(ws + (size_t)n1 * D_ENC * 2);         // n2*128*2 B
    float*  normM = (float*)(ws + (size_t)(n1 + n2) * D_ENC * 2);   // n1*4 B
    float*  normR = normM + n1;                                     // n2*4 B

    encode_kernel<<<n1 / 32, 256, 0, stream>>>(match, Wm, bm, mbf, normM);
    encode_kernel<<<n2 / 32, 256, 0, stream>>>(refer, Wr, br, rbf, normR);

    float* outM = (float*)d_out;
    float* outR = outM + n1;
    init_out_kernel<<<(n1 + n2 + 255) / 256, 256, 0, stream>>>(outM, n1 + n2);

    dim3 grid(n2 / 128, n1 / 256);
    pairwise_kernel<<<grid, 256, 0, stream>>>(mbf, rbf, normM, normR, outM, outR);

    finalize_kernel<<<(n1 + n2 + 255) / 256, 256, 0, stream>>>(outM, n1 + n2);
}
